// InstanceSegmentation_85469849191077
// MI455X (gfx1250) — matrix operand order repaired and checked by reading, not compiled, not run on hardware
//
#include <hip/hip_runtime.h>

typedef __attribute__((ext_vector_type(16))) _Float16 v16h;
typedef __attribute__((ext_vector_type(8)))  _Float16 v8h;
typedef __attribute__((ext_vector_type(8)))  float    v8f;

#define P_TOT 5456
#define NCLS  80
#define MAXI  100

// ---------------------------------------------------------------------------
// NCHW f32 -> NHWC f16 (feature maps)
// ---------------------------------------------------------------------------
__global__ void to_nhwc_f16_k(const float* __restrict__ in, _Float16* __restrict__ out,
                              int C, int HW, int total) {
    int idx = blockIdx.x * 256 + threadIdx.x;
    if (idx >= total) return;
    int p  = idx % HW;
    int bc = idx / HW;
    int c  = bc % C;
    int b  = bc / C;
    out[((size_t)b * HW + p) * C + c] = (_Float16)in[idx];
}

// ---------------------------------------------------------------------------
// OIHW f32 -> (9, Cout, Cin) f16 weight transform
// ---------------------------------------------------------------------------
__global__ void wtrans_k(const float* __restrict__ in, _Float16* __restrict__ out,
                         int Cout, int Cin, int total) {
    int i = blockIdx.x * 256 + threadIdx.x;
    if (i >= total) return;
    int c = i % Cin;
    int t = i / Cin;
    int n = t % Cout;
    int r = t / Cout;
    out[i] = (_Float16)in[((size_t)n * Cin + c) * 9 + r];
}

// ---------------------------------------------------------------------------
// bottom-head input: f3 + bilinear(f4) + bilinear(f5), emitted as NHWC f16
// ---------------------------------------------------------------------------
__device__ inline float bil_sample(const float* __restrict__ p, int S, int y, int x) {
    float scale = (float)S / 64.f;
    float fy = ((float)y + 0.5f) * scale - 0.5f;
    float fx = ((float)x + 0.5f) * scale - 0.5f;
    fy = fminf(fmaxf(fy, 0.f), (float)(S - 1));
    fx = fminf(fmaxf(fx, 0.f), (float)(S - 1));
    int y0 = (int)fy, x0 = (int)fx;
    int y1 = (y0 + 1 < S) ? y0 + 1 : S - 1;
    int x1 = (x0 + 1 < S) ? x0 + 1 : S - 1;
    float wy = fy - (float)y0, wx = fx - (float)x0;
    float v00 = p[y0 * S + x0], v01 = p[y0 * S + x1];
    float v10 = p[y1 * S + x0], v11 = p[y1 * S + x1];
    return v00 * (1.f - wy) * (1.f - wx) + v01 * (1.f - wy) * wx +
           v10 * wy * (1.f - wx) + v11 * wy * wx;
}

__global__ void fuse_bottom_k(const float* __restrict__ f3, const float* __restrict__ f4,
                              const float* __restrict__ f5, _Float16* __restrict__ out, int total) {
    int idx = blockIdx.x * 256 + threadIdx.x;   // idx = (b*256 + c)*4096 + p
    if (idx >= total) return;
    int p = idx & 4095;
    int x = p & 63, y = (p >> 6) & 63;
    int c = (idx >> 12) & 255;
    int b = idx >> 20;
    int bc = idx >> 12;
    float v = f3[idx];
    v += bil_sample(f4 + (size_t)bc * 1024, 32, y, x);
    v += bil_sample(f5 + (size_t)bc * 256, 16, y, x);
    out[((size_t)b * 4096 + p) * 256 + c] = (_Float16)v;
}

// ---------------------------------------------------------------------------
// implicit-GEMM conv3x3 (SAME) via V_WMMA_F32_16X16X32_F16
//   activations: NHWC f16, weights: (9, Cout, Cin) f16
//   block = 128 threads (4 waves); block tile = 64 pixels x 64 couts
//   per sync region each wave issues 4 WMMAs (4 M-subtiles, shared B frag)
//   grid = (ceil(HW/64), ceil(Cout/64), B)
// ---------------------------------------------------------------------------
__global__ __launch_bounds__(128)
void conv3x3_wmma_k(const _Float16* __restrict__ xt, const _Float16* __restrict__ wt,
                    const float* __restrict__ bias, float* __restrict__ out,
                    int H, int W, int Cin, int Cout, int outPixStride, int pixOff, int doRelu)
{
    __shared__ __align__(16) _Float16 sA[64 * 32];   // [m 0..63][k 0..31]
    const int tid  = threadIdx.x;
    const int lane = tid & 31;
    const int wv   = tid >> 5;
    const int HW   = H * W;
    const int m0   = blockIdx.x * 64;
    const int n0   = blockIdx.y * 64;
    const int b    = blockIdx.z;

    // A staging: thread -> (pixel row sm, 16-channel half kq)
    const int sm = tid >> 1;
    const int kq = (tid & 1) * 16;
    // fragment geometry (CDNA5 16x16x32 f16 layouts)
    const int row = lane & 15;
    const int hb  = (lane >> 4) * 8;    // A: elems 0-7 -> K hb..hb+7, 8-15 -> K 16+hb..
    const int kb  = (lane >> 4) * 16;   // B: same K order as A (elems 0-7 -> K hb..hb+7, 8-15 -> K 16+hb..)
    const int ng  = n0 + wv * 16 + (lane & 15);
    const bool nv = ng < Cout;
    const _Float16* wrow = wt + (size_t)ng * Cin + hb;

    v8f acc[4] = {};
    const v8h vz = {};

    for (int r = 0; r < 9; ++r) {
        const int dy = r / 3 - 1, dx = r % 3 - 1;
        const int p  = m0 + sm;
        const int y  = p / W + dy;
        const int x  = p % W + dx;
        const bool inb = (p < HW) && (y >= 0) && (y < H) && (x >= 0) && (x < W);
        const _Float16* aptr = inb ? (xt + (((size_t)b * H + y) * W + x) * Cin + kq) : (const _Float16*)0;
        const _Float16* bptr = nv ? (wrow + (size_t)r * Cout * Cin) : (const _Float16*)0;

        for (int c0 = 0; c0 < Cin; c0 += 32) {
            v8h a0 = vz, a1 = vz, b0 = vz, b1 = vz;
            if (inb) {                       // 32 contiguous channels of one pixel
                a0 = *(const v8h*)(aptr + c0);
                a1 = *(const v8h*)(aptr + c0 + 8);
            }
            if (nv) {                        // B fragment straight from global
                b0 = *(const v8h*)(bptr + c0);
                b1 = *(const v8h*)(bptr + c0 + 16);
            }
            *(v8h*)&sA[sm * 32 + kq]     = a0;
            *(v8h*)&sA[sm * 32 + kq + 8] = a1;
            __syncthreads();
            v16h bm = __builtin_shufflevector(b0, b1, 0,1,2,3,4,5,6,7,8,9,10,11,12,13,14,15);
            #pragma unroll
            for (int mt = 0; mt < 4; ++mt) {
                v8h alo = *(const v8h*)&sA[(mt * 16 + row) * 32 + hb];
                v8h ahi = *(const v8h*)&sA[(mt * 16 + row) * 32 + 16 + hb];
                v16h am = __builtin_shufflevector(alo, ahi, 0,1,2,3,4,5,6,7,8,9,10,11,12,13,14,15);
                acc[mt] = __builtin_amdgcn_wmma_f32_16x16x32_f16(false, am, false, bm,
                                                                 (short)0, acc[mt], false, false);
            }
            __syncthreads();
        }
    }
    // epilogue: D layout lane0-15 -> M=v, lane16-31 -> M=v+8; N = lane&15
    if (nv) {
        float bs = bias[ng];
        #pragma unroll
        for (int mt = 0; mt < 4; ++mt) {
            #pragma unroll
            for (int v = 0; v < 8; ++v) {
                int p = m0 + mt * 16 + v + 8 * (lane >> 4);
                if (p < HW) {
                    float val = acc[mt][v] + bs;
                    if (doRelu) val = fmaxf(val, 0.f);
                    out[((size_t)b * Cout + ng) * outPixStride + pixOff + p] = val;
                }
            }
        }
    }
}

// ---------------------------------------------------------------------------
// GroupNorm(32) + ReLU: channel-major f32 in -> NHWC f16 out
// grid = B*32 blocks, 256 threads
// ---------------------------------------------------------------------------
__global__ __launch_bounds__(256)
void groupnorm_relu_k(const float* __restrict__ in, const float* __restrict__ g,
                      const float* __restrict__ be, _Float16* __restrict__ outh,
                      int C, int HW)
{
    int b   = blockIdx.x >> 5;
    int grp = blockIdx.x & 31;
    int Cg  = C / 32;
    size_t base = ((size_t)b * C + (size_t)grp * Cg) * HW;
    int n = Cg * HW;
    float s = 0.f, s2 = 0.f;
    for (int i = threadIdx.x; i < n; i += 256) {
        float v = in[base + i];
        s += v; s2 += v * v;
    }
    __shared__ float rs[256], rq[256];
    rs[threadIdx.x] = s; rq[threadIdx.x] = s2;
    __syncthreads();
    for (int o = 128; o > 0; o >>= 1) {
        if (threadIdx.x < o) { rs[threadIdx.x] += rs[threadIdx.x + o]; rq[threadIdx.x] += rq[threadIdx.x + o]; }
        __syncthreads();
    }
    float mean = rs[0] / (float)n;
    float var  = rq[0] / (float)n - mean * mean;
    float inv  = rsqrtf(var + 1e-5f);
    for (int i = threadIdx.x; i < n; i += 256) {
        int cl = i / HW;
        int p  = i - cl * HW;
        int c  = grp * Cg + cl;
        float v = (in[base + i] - mean) * inv * g[c] + be[c];
        outh[((size_t)b * HW + p) * C + c] = (_Float16)fmaxf(v, 0.f);
    }
}

// ---------------------------------------------------------------------------
// per-candidate score (sigmoid of max logit), argmax class, decoded box,
// normalized centers
// ---------------------------------------------------------------------------
__device__ inline void level_info(int p, int& poff, int& dim, float& stride) {
    if      (p < 4096) { poff = 0;    dim = 64; stride = 8.f;   }
    else if (p < 5120) { poff = 4096; dim = 32; stride = 16.f;  }
    else if (p < 5376) { poff = 5120; dim = 16; stride = 32.f;  }
    else if (p < 5440) { poff = 5376; dim = 8;  stride = 64.f;  }
    else               { poff = 5440; dim = 4;  stride = 128.f; }
}

__global__ void score_box_k(const float* __restrict__ logits, const float* __restrict__ boxh,
                            float* __restrict__ scores, int* __restrict__ classes,
                            float* __restrict__ boxes, float* __restrict__ centersN, int B)
{
    int idx = blockIdx.x * 256 + threadIdx.x;
    if (idx >= B * P_TOT) return;
    int p = idx % P_TOT, b = idx / P_TOT;
    float mx = -3.4e38f; int am = 0;
    for (int c = 0; c < NCLS; ++c) {
        float v = logits[((size_t)b * NCLS + c) * P_TOT + p];
        if (v > mx) { mx = v; am = c; }
    }
    scores[idx]  = 1.f / (1.f + expf(-mx));
    classes[idx] = am;
    int poff, dim; float stride;
    level_info(p, poff, dim, stride);
    int pl = p - poff;
    int yy = pl / dim, xx = pl % dim;
    float cx = ((float)xx + 0.5f) * stride;
    float cy = ((float)yy + 0.5f) * stride;
    float b0 = boxh[((size_t)b * 4 + 0) * P_TOT + p];
    float b1 = boxh[((size_t)b * 4 + 1) * P_TOT + p];
    float b2 = boxh[((size_t)b * 4 + 2) * P_TOT + p];
    float b3 = boxh[((size_t)b * 4 + 3) * P_TOT + p];
    boxes[(size_t)idx * 4 + 0] = cx - (stride * b0 + 1.f);
    boxes[(size_t)idx * 4 + 1] = cy - (stride * b1 + 1.f);
    boxes[(size_t)idx * 4 + 2] = cx + (stride * b2 + 1.f);
    boxes[(size_t)idx * 4 + 3] = cy + (stride * b3 + 1.f);
    if (b == 0) {
        centersN[p * 2 + 0] = cx / ((float)dim * stride);
        centersN[p * 2 + 1] = cy / ((float)dim * stride);
    }
}

// ---------------------------------------------------------------------------
// greedy NMS, one block per batch
// ---------------------------------------------------------------------------
__global__ __launch_bounds__(256)
void nms_k(const float* __restrict__ scores, const int* __restrict__ classes,
           const float* __restrict__ boxes, const float* __restrict__ centersN,
           const float* __restrict__ dyn, float* __restrict__ swork,
           float* __restrict__ out, float* __restrict__ selParams,
           float* __restrict__ selCenters, int* __restrict__ selValid)
{
    int b = blockIdx.x, t = threadIdx.x;
    float* sw = swork + (size_t)b * P_TOT;
    const float* bx = boxes + (size_t)b * P_TOT * 4;
    const float NEGINF = -__builtin_inff();
    for (int i = t; i < P_TOT; i += 256) {
        float s = scores[b * P_TOT + i];
        sw[i] = (s > 0.05f) ? s : NEGINF;
    }
    __syncthreads();
    __shared__ float rv[256];
    __shared__ int   ri[256];
    for (int it = 0; it < MAXI; ++it) {
        float best = NEGINF; int bi = 0;
        for (int i = t; i < P_TOT; i += 256) {
            float v = sw[i];
            if (v > best) { best = v; bi = i; }
        }
        rv[t] = best; ri[t] = bi;
        __syncthreads();
        for (int o = 128; o > 0; o >>= 1) {
            if (t < o && rv[t + o] > rv[t]) { rv[t] = rv[t + o]; ri[t] = ri[t + o]; }
            __syncthreads();
        }
        int j = ri[0]; float ps = rv[0];
        int valid = (ps > 0.05f) ? 1 : 0;
        if (t == 0) {
            out[2 + b * MAXI + it]            = valid ? ps : 0.f;
            out[2 + 2 * MAXI + b * MAXI + it] = valid ? (float)classes[b * P_TOT + j] : 0.f;
            selValid[b * MAXI + it] = valid;
            selCenters[(b * MAXI + it) * 2 + 0] = valid ? centersN[j * 2 + 0] : 0.f;
            selCenters[(b * MAXI + it) * 2 + 1] = valid ? centersN[j * 2 + 1] : 0.f;
        }
        for (int c = t; c < 169; c += 256)
            selParams[((size_t)b * MAXI + it) * 169 + c] = dyn[((size_t)b * 169 + c) * P_TOT + j];
        __syncthreads();
        float jx1 = bx[j * 4 + 0], jy1 = bx[j * 4 + 1];
        float jx2 = bx[j * 4 + 2], jy2 = bx[j * 4 + 3];
        float ja = (jx2 - jx1) * (jy2 - jy1);
        for (int i = t; i < P_TOT; i += 256) {
            float x1 = fmaxf(jx1, bx[i * 4 + 0]), y1 = fmaxf(jy1, bx[i * 4 + 1]);
            float x2 = fminf(jx2, bx[i * 4 + 2]), y2 = fminf(jy2, bx[i * 4 + 3]);
            float inter = fmaxf(x2 - x1, 0.f) * fmaxf(y2 - y1, 0.f);
            float a2 = (bx[i * 4 + 2] - bx[i * 4 + 0]) * (bx[i * 4 + 3] - bx[i * 4 + 1]);
            float iou = inter / (ja + a2 - inter + 1e-9f);
            if (iou > 0.6f) sw[i] = NEGINF;
        }
        __syncthreads();
        if (t == 0) sw[j] = NEGINF;
        __syncthreads();
    }
    if (t == 0) {
        int cnt = 0;
        for (int i = 0; i < MAXI; ++i)
            if (out[2 + b * MAXI + i] > 0.5f) cnt++;
        out[b] = (float)cnt;
    }
}

// ---------------------------------------------------------------------------
// dynamic mask head: 10->8->8->1 pointwise convs over 64x64 per instance
// ---------------------------------------------------------------------------
__global__ __launch_bounds__(256)
void mask_k(const float* __restrict__ fbot, const float* __restrict__ selParams,
            const float* __restrict__ selCenters, const int* __restrict__ selValid,
            float* __restrict__ out)
{
    int bi = blockIdx.x;
    int b = bi / MAXI;
    __shared__ float prm[169];
    for (int i = threadIdx.x; i < 169; i += 256) prm[i] = selParams[(size_t)bi * 169 + i];
    __syncthreads();
    int valid = selValid[bi];
    float cxp = selCenters[bi * 2 + 0] * 64.f;
    float cyp = selCenters[bi * 2 + 1] * 64.f;
    float* mo = out + 2 + 4 * MAXI + (size_t)bi * 4096;
    const float* fb = fbot + (size_t)b * 8 * 4096;
    for (int p = threadIdx.x; p < 4096; p += 256) {
        int y = p >> 6, x = p & 63;
        float xin[10];
        xin[0] = ((float)x - cxp) / 64.f;
        xin[1] = ((float)y - cyp) / 64.f;
        #pragma unroll
        for (int c = 0; c < 8; ++c) xin[2 + c] = fb[c * 4096 + p];
        float h1[8], h2[8];
        #pragma unroll
        for (int o = 0; o < 8; ++o) {
            float s = prm[80 + o];
            #pragma unroll
            for (int c = 0; c < 10; ++c) s += prm[o * 10 + c] * xin[c];
            h1[o] = fmaxf(s, 0.f);
        }
        #pragma unroll
        for (int o = 0; o < 8; ++o) {
            float s = prm[152 + o];
            #pragma unroll
            for (int c = 0; c < 8; ++c) s += prm[88 + o * 8 + c] * h1[c];
            h2[o] = fmaxf(s, 0.f);
        }
        float s = prm[168];
        #pragma unroll
        for (int c = 0; c < 8; ++c) s += prm[160 + c] * h2[c];
        float m = 1.f / (1.f + expf(-s));
        mo[p] = valid ? m : 0.f;
    }
}

// ---------------------------------------------------------------------------
// host orchestration
// ---------------------------------------------------------------------------
extern "C" void kernel_launch(void* const* d_in, const int* in_sizes, int n_in,
                              void* d_out, int out_size, void* d_ws, size_t ws_size,
                              hipStream_t stream)
{
    (void)in_sizes; (void)n_in; (void)out_size; (void)ws_size;
    char* ws = (char*)d_ws;
    size_t off = 0;
    auto alloc = [&](size_t bytes) -> char* {
        off = (off + 255) & ~(size_t)255;
        char* p = ws + off;
        off += bytes;
        return p;
    };
    const int B = 2;
    const int HWs[5]  = {4096, 1024, 256, 64, 16};
    const int DIMs[5] = {64, 32, 16, 8, 4};
    const int POFF[5] = {0, 4096, 5120, 5376, 5440};

    _Float16* feat16[5];
    for (int l = 0; l < 5; ++l) feat16[l] = (_Float16*)alloc((size_t)B * 256 * HWs[l] * 2);
    _Float16* fbin16 = (_Float16*)alloc((size_t)B * 256 * 4096 * 2);
    float*    tmpf   = (float*)alloc((size_t)B * 256 * 4096 * 4);
    _Float16* act16  = (_Float16*)alloc((size_t)B * 256 * 4096 * 2);
    float* logits    = (float*)alloc((size_t)B * NCLS * P_TOT * 4);
    float* boxh      = (float*)alloc((size_t)B * 4 * P_TOT * 4);
    float* ctrlo     = (float*)alloc((size_t)B * 169 * P_TOT * 4);
    float* fbot      = (float*)alloc((size_t)B * 8 * 4096 * 4);
    float* scores    = (float*)alloc((size_t)B * P_TOT * 4);
    int*   classes   = (int*)alloc((size_t)B * P_TOT * 4);
    float* boxes     = (float*)alloc((size_t)B * P_TOT * 4 * 4);
    float* centersN  = (float*)alloc((size_t)P_TOT * 2 * 4);
    float* swork     = (float*)alloc((size_t)B * P_TOT * 4);
    float* selParams = (float*)alloc((size_t)B * MAXI * 169 * 4);
    float* selCenters= (float*)alloc((size_t)B * MAXI * 2 * 4);
    int*   selValid  = (int*)alloc((size_t)B * MAXI * 4);

    // features -> NHWC f16
    for (int l = 0; l < 5; ++l) {
        int total = B * 256 * HWs[l];
        to_nhwc_f16_k<<<dim3((total + 255) / 256), dim3(256), 0, stream>>>(
            (const float*)d_in[l], feat16[l], 256, HWs[l], total);
    }
    // bottom input fuse (NHWC f16)
    {
        int total = B * 256 * 4096;
        fuse_bottom_k<<<dim3((total + 255) / 256), dim3(256), 0, stream>>>(
            (const float*)d_in[0], (const float*)d_in[1], (const float*)d_in[2], fbin16, total);
    }

    // heads in d_in (alphabetical jax-tree order): bottom, box, class, ctrl
    // per head: L blocks of (b, be, g, w), then fb, fw
    const int baseIdx[4] = {5, 19, 37, 55};
    const int Lh[4]      = {3, 4, 4, 4};
    const int cmid[4]    = {128, 256, 256, 256};
    const int coutF[4]   = {8, 4, NCLS, 169};

    _Float16* wblk[4][4];
    _Float16* wfin[4];
    for (int h = 0; h < 4; ++h) {
        int cin = 256;
        for (int i = 0; i < Lh[h]; ++i) {
            int total = 9 * cmid[h] * cin;
            wblk[h][i] = (_Float16*)alloc((size_t)total * 2);
            wtrans_k<<<dim3((total + 255) / 256), dim3(256), 0, stream>>>(
                (const float*)d_in[baseIdx[h] + i * 4 + 3], wblk[h][i], cmid[h], cin, total);
            cin = cmid[h];
        }
        int totf = 9 * coutF[h] * cmid[h];
        wfin[h] = (_Float16*)alloc((size_t)totf * 2);
        wtrans_k<<<dim3((totf + 255) / 256), dim3(256), 0, stream>>>(
            (const float*)d_in[baseIdx[h] + Lh[h] * 4 + 1], wfin[h], coutF[h], cmid[h], totf);
    }

    auto runConv = [&](const _Float16* x, const _Float16* w, const float* bias, float* o,
                       int H, int W, int Cin, int Cout, int pixStride, int pixOff, int relu) {
        dim3 grid((H * W + 63) / 64, (Cout + 63) / 64, B);
        conv3x3_wmma_k<<<grid, dim3(128), 0, stream>>>(x, w, bias, o, H, W, Cin, Cout,
                                                       pixStride, pixOff, relu);
    };
    auto runGN = [&](const float* in, const float* g, const float* be, _Float16* o, int C, int HW) {
        groupnorm_relu_k<<<dim3(B * 32), dim3(256), 0, stream>>>(in, g, be, o, C, HW);
    };

    // class/box/ctrl heads over 5 levels
    for (int h = 1; h <= 3; ++h) {
        float* hout = (h == 1) ? boxh : (h == 2) ? logits : ctrlo;
        for (int l = 0; l < 5; ++l) {
            int Hl = DIMs[l], Wl = DIMs[l], HW = HWs[l];
            const _Float16* cur = feat16[l];
            int cin = 256;
            for (int i = 0; i < Lh[h]; ++i) {
                runConv(cur, wblk[h][i], (const float*)d_in[baseIdx[h] + i * 4 + 0],
                        tmpf, Hl, Wl, cin, cmid[h], HW, 0, 0);
                runGN(tmpf, (const float*)d_in[baseIdx[h] + i * 4 + 2],
                      (const float*)d_in[baseIdx[h] + i * 4 + 1], act16, cmid[h], HW);
                cur = act16; cin = cmid[h];
            }
            runConv(cur, wfin[h], (const float*)d_in[baseIdx[h] + Lh[h] * 4 + 0],
                    hout, Hl, Wl, cmid[h], coutF[h], P_TOT, POFF[l], (h == 1) ? 1 : 0);
        }
    }
    // bottom head (64x64 only)
    {
        const _Float16* cur = fbin16;
        int cin = 256;
        for (int i = 0; i < Lh[0]; ++i) {
            runConv(cur, wblk[0][i], (const float*)d_in[baseIdx[0] + i * 4 + 0],
                    tmpf, 64, 64, cin, 128, 4096, 0, 0);
            runGN(tmpf, (const float*)d_in[baseIdx[0] + i * 4 + 2],
                  (const float*)d_in[baseIdx[0] + i * 4 + 1], act16, 128, 4096);
            cur = act16; cin = 128;
        }
        runConv(cur, wfin[0], (const float*)d_in[baseIdx[0] + Lh[0] * 4 + 0],
                fbot, 64, 64, 128, 8, 4096, 0, 0);
    }

    score_box_k<<<dim3((B * P_TOT + 255) / 256), dim3(256), 0, stream>>>(
        logits, boxh, scores, classes, boxes, centersN, B);

    nms_k<<<dim3(B), dim3(256), 0, stream>>>(scores, classes, boxes, centersN, ctrlo,
                                             swork, (float*)d_out, selParams, selCenters, selValid);

    mask_k<<<dim3(B * MAXI), dim3(256), 0, stream>>>(fbot, selParams, selCenters, selValid,
                                                     (float*)d_out);
}
